// SelfAttention1D_48962627174948
// MI455X (gfx1250) — matrix operand order repaired and checked by reading, not compiled, not run on hardware
//
#include <hip/hip_runtime.h>
#include <stdint.h>

typedef __bf16 bf16;
typedef uint32_t u32;
typedef __attribute__((ext_vector_type(16))) __bf16 v16bf;
typedef __attribute__((ext_vector_type(8)))  __bf16 v8bf;
typedef __attribute__((ext_vector_type(8)))  float  v8f;
typedef __attribute__((ext_vector_type(4)))  u32    u32x4;
typedef __attribute__((ext_vector_type(8)))  u32    u32x8;

// B=4, N=4096, C=512, C8=64 (fixed by the reference)
#define BATCH 4
#define NSEQ  4096
#define CDIM  512
#define C8    64
#define ROWS  (BATCH * NSEQ)   // 16384 flat rows

union FragA { v16bf v; v8bf h[2]; };

// A-fragment (16x32, 16-bit): lane half h holds K in {h*16..h*16+15}
__device__ inline v16bf load_a_bf16(const bf16* row, int half) {
  FragA fr;
  fr.h[0] = *(const v8bf*)(row + half * 16);
  fr.h[1] = *(const v8bf*)(row + half * 16 + 8);
  return fr.v;
}

__device__ inline v16bf load_a_f32(const float* row, int half) {
  FragA fr;
  const float* p0 = row + half * 16;
  const float* p1 = row + half * 16 + 8;
#pragma unroll
  for (int i = 0; i < 8; i++) {
    fr.h[0][i] = (bf16)p0[i];
    fr.h[1][i] = (bf16)p1[i];
  }
  return fr.v;
}

__device__ inline v8f wmma_bf16(v16bf a, v16bf b, v8f c) {
  return __builtin_amdgcn_wmma_f32_16x16x32_bf16(
      false, a, false, b, (short)0, c, false, false);
}

// Single-instruction lane^MASK exchange: ds_swizzle_b32, group-of-32 mode,
// offset = {xor_mask<<10 | or_mask<<5 | and_mask}; and=0x1f, or=0, xor=MASK.
template <int MASK>
__device__ inline float swz_xor(float v) {
  int r = __builtin_amdgcn_ds_swizzle(__float_as_int(v), 0x1f | (MASK << 10));
  return __int_as_float(r);
}

// Butterfly reductions across the 16-lane half (rows of the C-layout tile).
__device__ inline float rowmax16(float v) {
  v = fmaxf(v, swz_xor<1>(v));
  v = fmaxf(v, swz_xor<2>(v));
  v = fmaxf(v, swz_xor<4>(v));
  v = fmaxf(v, swz_xor<8>(v));
  return v;
}
__device__ inline float rowsum16(float v) {
  v += swz_xor<1>(v);
  v += swz_xor<2>(v);
  v += swz_xor<4>(v);
  v += swz_xor<8>(v);
  return v;
}

// TDM: DMA a 2D bf16 tile (tile0 x tile1 elems) from global into LDS.
// D# built per CDNA5 ISA; 2-group form (<=2D).
__device__ inline void tdm_load_2d(u32 lds_byte_off, const void* gaddr,
                                   u32 tdim0, u32 tdim1, u32 tile0, u32 tile1,
                                   u32 stride0) {
  const uint64_t ga = (uint64_t)(uintptr_t)gaddr;
  u32x4 g0 = {1u,                               // count=1 (valid user D#)
              lds_byte_off,                      // lds_addr
              (u32)ga,                           // global_addr[31:0]
              (u32)(ga >> 32) | (2u << 30)};     // global_addr[56:32]|type=2
  u32x8 g1 = {(1u << 16),                        // data_size=1 -> 2B elems
              (tdim0 & 0xFFFFu) << 16,           // tensor_dim0[15:0]
              (tdim0 >> 16) | ((tdim1 & 0xFFFFu) << 16),
              (tdim1 >> 16) | (tile0 << 16),     // tile_dim0
              tile1,                             // tile_dim1 (tile_dim2=0)
              stride0,                           // tensor_dim0_stride[31:0]
              0u, 0u};
  asm volatile("tensor_load_to_lds %0, %1" ::"s"(g0), "s"(g1) : "memory");
}

// ---------------------------------------------------------------------------
// Phase 1: bf16 projection GEMM  out = x @ W + bias  (optionally transposed)
// Block: 256 thr (8 waves). Block tile: 128 rows x 64 cols. Wave: 16 rows.
// ---------------------------------------------------------------------------
template <int NCOLS, bool TRANSPOSED>
__global__ __launch_bounds__(256) void proj_kernel(
    const float* __restrict__ x, const float* __restrict__ W,
    const float* __restrict__ bias, bf16* __restrict__ out) {
  __shared__ alignas(64) bf16 ldsW[64 * 32];  // [c][k] transposed W tile

  const int tid  = threadIdx.x;
  const int wave = tid >> 5, lane = tid & 31;
  const int half = lane >> 4, l16 = lane & 15;
  const int r    = blockIdx.x * 128 + wave * 16;
  const int c0   = blockIdx.y * 64;

  v8f acc[4];
#pragma unroll
  for (int t = 0; t < 4; t++)
#pragma unroll
    for (int j = 0; j < 8; j++) acc[t][j] = 0.f;

  const int arow = r + l16;

  for (int k0 = 0; k0 < CDIM; k0 += 32) {
    __syncthreads();
    {  // stage W[k0..k0+31, c0..c0+63] into LDS as [c][k] bf16
      int i  = tid * 8;  // 2048 elems / 256 threads
      int kk = i >> 6;
      int cc = i & 63;
      const float* wp = W + (size_t)(k0 + kk) * NCOLS + (c0 + cc);
#pragma unroll
      for (int e = 0; e < 8; e++) ldsW[(cc + e) * 32 + kk] = (bf16)wp[e];
    }
    __syncthreads();

    v16bf a = load_a_f32(x + (size_t)arow * CDIM + k0, half);
#pragma unroll
    for (int t = 0; t < 4; t++) {
      v16bf b = *(const v16bf*)&ldsW[(t * 16 + l16) * 32 + half * 16];
      acc[t] = wmma_bf16(a, b, acc[t]);
    }
  }

#pragma unroll
  for (int t = 0; t < 4; t++) {
    int   col = c0 + t * 16 + l16;
    float bv  = bias[col];
#pragma unroll
    for (int j = 0; j < 8; j++) {
      int   row = r + j + half * 8;
      float v   = acc[t][j] + bv;
      if constexpr (!TRANSPOSED) {
        out[(size_t)row * NCOLS + col] = (bf16)v;
      } else {  // hT[b, c, n]
        int b = row >> 12, n = row & (NSEQ - 1);
        out[((size_t)b * CDIM + col) * NSEQ + n] = (bf16)v;
      }
    }
  }
}

// ---------------------------------------------------------------------------
// Phase 2: flash attention.  Block: 256 thr (8 waves), one 16-query tile.
// g key-tiles (shared by all 8 waves) are DMA'd into LDS by the TDM, double
// buffered, 1 barrier/iter.  hT fragments load direct (contiguous per lane)
// and are hoisted above the softmax so VALU work hides their latency.
// ---------------------------------------------------------------------------
__global__ __launch_bounds__(256) void attn_kernel(
    const float* __restrict__ x, const bf16* __restrict__ f,
    const bf16* __restrict__ g, const bf16* __restrict__ hT,
    const float* __restrict__ gamma_p, float* __restrict__ out) {
  __shared__ alignas(64) bf16 ldsG[2][32 * 64];   // double-buffered key tile
  __shared__ alignas(64) bf16 ldsP[8 * 16 * 32];  // per-wave 16x32 P tile

  const int tid  = threadIdx.x;
  const int wave = tid >> 5, lane = tid & 31;
  const int half = lane >> 4, l16 = lane & 15;

  const int b   = blockIdx.x >> 8;
  const int q0  = (blockIdx.x & 255) * 16;
  const int gr0 = b * NSEQ + q0;  // flat row base in [B*N]
  const int c0  = wave * 64;
  bf16* myP = ldsP + wave * 512;

  // Q fragments: f[gr0+l16, 0..63], kept in registers for all 128 key tiles
  const bf16* frow = f + (size_t)(gr0 + l16) * C8;
  v16bf aQlo = load_a_bf16(frow, half);
  v16bf aQhi = load_a_bf16(frow + 32, half);

  v8f   acc[4];
  float m[8], l[8];
#pragma unroll
  for (int j = 0; j < 8; j++) { m[j] = -1e30f; l[j] = 0.f; }
#pragma unroll
  for (int t = 0; t < 4; t++)
#pragma unroll
    for (int j = 0; j < 8; j++) acc[t][j] = 0.f;

  const bf16* gB = g + (size_t)b * NSEQ * C8;
  const bf16* hB = hT + (size_t)b * CDIM * NSEQ;

  // Prologue: DMA key tile 0 into buffer 0
  if (wave == 0) {
    tdm_load_2d((u32)(uintptr_t)&ldsG[0][0], gB, C8, NSEQ, C8, 32, C8);
    __builtin_amdgcn_s_wait_tensorcnt(0);
  }
  __syncthreads();

  for (int it = 0; it < NSEQ / 32; ++it) {
    const int kb  = it * 32;
    const int cur = it & 1;

    // Kick off DMA of the next key tile into the other buffer
    if (wave == 0 && it + 1 < NSEQ / 32) {
      tdm_load_2d((u32)(uintptr_t)&ldsG[cur ^ 1][0],
                  gB + (size_t)(kb + 32) * C8, C8, NSEQ, C8, 32, C8);
    }

    // ---- S = Q @ K^T for two 16-key groups (unscaled, per reference) ----
    const bf16* gt = ldsG[cur];
    v8f S[2];
#pragma unroll
    for (int s = 0; s < 2; s++) {
      const bf16* grow = gt + (s * 16 + l16) * C8;
      v16bf bglo = *(const v16bf*)(grow + half * 16);
      v16bf bghi = *(const v16bf*)(grow + 32 + half * 16);
      v8f z;
#pragma unroll
      for (int j = 0; j < 8; j++) z[j] = 0.f;
      z    = wmma_bf16(aQlo, bglo, z);
      S[s] = wmma_bf16(aQhi, bghi, z);
    }

    // ---- hoist hT fragment loads: softmax VALU work hides their latency ----
    v16bf bh[4];
#pragma unroll
    for (int t = 0; t < 4; t++) {
      const bf16* hrow =
          hB + (size_t)(c0 + t * 16 + l16) * NSEQ + kb + half * 16;
      bh[t] = *(const v16bf*)hrow;
    }

    // ---- online softmax; emit P (bf16) into per-wave LDS ----
#pragma unroll
    for (int j = 0; j < 8; j++) {
      float rmax  = rowmax16(fmaxf(S[0][j], S[1][j]));
      float mnew  = fmaxf(m[j], rmax);
      float scale = __expf(m[j] - mnew);
      float p0    = __expf(S[0][j] - mnew);
      float p1    = __expf(S[1][j] - mnew);
      l[j] = l[j] * scale + rowsum16(p0 + p1);
      m[j] = mnew;
#pragma unroll
      for (int t = 0; t < 4; t++) acc[t][j] *= scale;
      int q = j + half * 8;
      myP[q * 32 + l16]      = (bf16)p0;
      myP[q * 32 + l16 + 16] = (bf16)p1;
    }

    // ---- acc += P(16x32) @ h(32x16) per 16-channel group ----
    v16bf aP = load_a_bf16(myP + l16 * 32, half);
#pragma unroll
    for (int t = 0; t < 4; t++) acc[t] = wmma_bf16(aP, bh[t], acc[t]);

    // Fence the in-flight DMA, then release both buffers for next iter
    if (wave == 0) __builtin_amdgcn_s_wait_tensorcnt(0);
    __syncthreads();
  }

  // ---- epilogue: out = x + gamma * acc / l ----
  const float gamma = *gamma_p;
  float invl[8];
#pragma unroll
  for (int j = 0; j < 8; j++) invl[j] = 1.f / l[j];
#pragma unroll
  for (int t = 0; t < 4; t++) {
#pragma unroll
    for (int j = 0; j < 8; j++) {
      int    row = gr0 + j + half * 8;
      int    col = c0 + t * 16 + l16;
      size_t idx = (size_t)row * CDIM + col;
      out[idx] = x[idx] + gamma * (acc[t][j] * invl[j]);
    }
  }
}

extern "C" void kernel_launch(void* const* d_in, const int* in_sizes, int n_in,
                              void* d_out, int out_size, void* d_ws,
                              size_t ws_size, hipStream_t stream) {
  const float* x     = (const float*)d_in[0];
  const float* Wf    = (const float*)d_in[1];
  const float* bf_   = (const float*)d_in[2];
  const float* Wg    = (const float*)d_in[3];
  const float* bg    = (const float*)d_in[4];
  const float* Wh    = (const float*)d_in[5];
  const float* bh    = (const float*)d_in[6];
  const float* gamma = (const float*)d_in[7];
  float* out = (float*)d_out;

  // workspace: f (2MB) | g (2MB) | hT (16MB), all bf16
  bf16* fbuf  = (bf16*)d_ws;
  bf16* gbuf  = fbuf + (size_t)ROWS * C8;
  bf16* hTbuf = gbuf + (size_t)ROWS * C8;

  dim3 blk(256);
  proj_kernel<C8, false><<<dim3(ROWS / 128, 1), blk, 0, stream>>>(x, Wf, bf_,
                                                                  fbuf);
  proj_kernel<C8, false><<<dim3(ROWS / 128, 1), blk, 0, stream>>>(x, Wg, bg,
                                                                  gbuf);
  proj_kernel<CDIM, true><<<dim3(ROWS / 128, CDIM / 64), blk, 0, stream>>>(
      x, Wh, bh, hTbuf);
  attn_kernel<<<dim3(BATCH * NSEQ / 16), blk, 0, stream>>>(x, fbuf, gbuf,
                                                           hTbuf, gamma, out);
}
